// GAT_42485816492678
// MI455X (gfx1250) — compile-verified, matrix operand order checked by reading, not run on hardware
//
#include <hip/hip_runtime.h>
#include <hip/hip_bf16.h>

#define NNODES 100000
#define HCDIM  64
#define GGR    64
#define NEG_SLOPE 0.2f

typedef __attribute__((ext_vector_type(16))) __bf16 v16bf;
typedef __attribute__((ext_vector_type(8)))  __bf16 v8bf;
typedef __attribute__((ext_vector_type(8)))  float  v8f;

// ---------- helpers ----------
__device__ inline __bf16 f2bf(float f) {
    unsigned u = __float_as_uint(f);
    unsigned r = (u + 0x7FFFu + ((u >> 16) & 1u)) >> 16;   // RNE
    return __builtin_bit_cast(__bf16, (unsigned short)r);
}
// monotonic float<->uint mapping for atomic max on signed floats
__device__ inline unsigned fmap(float f) {
    unsigned u = __float_as_uint(f);
    return (u >> 31) ? ~u : (u | 0x80000000u);
}
__device__ inline float funmap(unsigned u) {
    return (u >> 31) ? __uint_as_float(u & 0x7FFFFFFFu) : __uint_as_float(~u);
}

// ---------- f32 -> bf16 bulk convert (4 elements/thread, 16B load / 8B store) ----------
__global__ void cvt_bf16x4(const float* __restrict__ src, __bf16* __restrict__ dst, int n4) {
    int i = blockIdx.x * blockDim.x + threadIdx.x;
    if (i >= n4) return;
    float4 v = ((const float4*)src)[i];
    union { __bf16 e[4]; unsigned long long u; } p;
    p.e[0] = f2bf(v.x); p.e[1] = f2bf(v.y); p.e[2] = f2bf(v.z); p.e[3] = f2bf(v.w);
    ((unsigned long long*)dst)[i] = p.u;
}

// ---------- WMMA GEMM: HL[N,64] = X[N,K] @ W[64,K]^T  (bf16 in, f32 acc) ----------
// grid.x = N/16 (exact), block = 128 (4 waves; wave w -> output cols [16w,16w+16))
// K templated so the loop fully unrolls into back-to-back v_wmma ops.
template <int KD>
__global__ void gat_gemm_wmma(const __bf16* __restrict__ X, const __bf16* __restrict__ W,
                              float* __restrict__ HL) {
    const int lane  = threadIdx.x & 31;
    const int wave  = threadIdx.x >> 5;
    const int mbase = blockIdx.x * 16;
    const int nbase = wave * 16;
    const int lh = lane & 15;             // row (A) / col (B) within tile
    const int kg = (lane >> 4) << 3;      // K sub-group: 0 or 8
    const int mo = (lane >> 4) << 3;      // C/D: lanes 16-31 hold M=8..15

    const __bf16* __restrict__ xrow = X + (size_t)(mbase + lh) * KD + kg;
    const __bf16* __restrict__ wrow = W + (size_t)(nbase + lh) * KD + kg;

    v8f acc = {};
    #pragma unroll
    for (int k0 = 0; k0 < KD; k0 += 32) {
        // ISA 16-bit operand layout: VGPR0-3 = K[kg..kg+7], VGPR4-7 = K[kg+16..kg+23]
        union { v16bf v; v8bf h[2]; } a, b;
        a.h[0] = *(const v8bf*)(xrow + k0);
        a.h[1] = *(const v8bf*)(xrow + k0 + 16);
        b.h[0] = *(const v8bf*)(wrow + k0);
        b.h[1] = *(const v8bf*)(wrow + k0 + 16);
        acc = __builtin_amdgcn_wmma_f32_16x16x32_bf16(
                  false, a.v, false, b.v, (short)0, acc, false, false);
    }
    #pragma unroll
    for (int r = 0; r < 8; ++r) {
        int m = mbase + mo + r;
        HL[(size_t)m * HCDIM + nbase + lh] = acc[r];
    }
}

// ---------- per-node attention coefficients ----------
__global__ void gat_alpha(const float* __restrict__ HL, const float* __restrict__ avs,
                          const float* __restrict__ avd,
                          float* __restrict__ as, float* __restrict__ ad) {
    int n = blockIdx.x * blockDim.x + threadIdx.x;
    if (n >= NNODES) return;
    const float* h = HL + (size_t)n * HCDIM;
    #pragma unroll
    for (int hd = 0; hd < 2; ++hd) {
        float s0 = 0.f, s1 = 0.f;
        #pragma unroll 8
        for (int c = 0; c < 32; ++c) {
            float v = h[hd * 32 + c];
            s0 += v * avs[hd * 32 + c];
            s1 += v * avd[hd * 32 + c];
        }
        as[n * 2 + hd] = s0;
        ad[n * 2 + hd] = s1;
    }
}

__global__ void gat_init_ms(unsigned* __restrict__ mu, float* __restrict__ ss) {
    int i = blockIdx.x * blockDim.x + threadIdx.x;
    if (i < NNODES * 2) { mu[i] = 0x007FFFFFu; /* fmap(-inf) */ ss[i] = 0.f; }
}

// ---------- pass 1: segment max ----------
__global__ void gat_edge_max(const int* __restrict__ ei, int E, int Etot,
                             const float* __restrict__ as, const float* __restrict__ ad,
                             unsigned* __restrict__ mu) {
    int idx = blockIdx.x * 256 + threadIdx.x;
    if (idx >= Etot) return;
    __builtin_prefetch(ei + idx + 8192, 0, 0);   // stream edge list through GL2
    int s, d;
    if (idx < E) { s = ei[idx]; d = ei[E + idx]; } else { s = d = idx - E; }
    #pragma unroll
    for (int h = 0; h < 2; ++h) {
        float e = as[s * 2 + h] + ad[d * 2 + h];
        e = e > 0.f ? e : NEG_SLOPE * e;
        atomicMax(&mu[d * 2 + h], fmap(e));
    }
}

// ---------- pass 2: exp-sum ----------
__global__ void gat_edge_sum(const int* __restrict__ ei, int E, int Etot,
                             const float* __restrict__ as, const float* __restrict__ ad,
                             const unsigned* __restrict__ mu, float* __restrict__ ss) {
    int idx = blockIdx.x * 256 + threadIdx.x;
    if (idx >= Etot) return;
    __builtin_prefetch(ei + idx + 8192, 0, 0);
    int s, d;
    if (idx < E) { s = ei[idx]; d = ei[E + idx]; } else { s = d = idx - E; }
    #pragma unroll
    for (int h = 0; h < 2; ++h) {
        float e = as[s * 2 + h] + ad[d * 2 + h];
        e = e > 0.f ? e : NEG_SLOPE * e;
        float p = __expf(e - funmap(mu[d * 2 + h]));
        atomicAdd(&ss[d * 2 + h], p);
    }
}

// ---------- pass 3: weighted message scatter (16 lanes per edge, 4 ch/lane) ----------
__global__ void gat_edge_msg(const int* __restrict__ ei, int E, int Etot,
                             const float* __restrict__ HL,
                             const float* __restrict__ as, const float* __restrict__ ad,
                             const unsigned* __restrict__ mu, const float* __restrict__ ss,
                             float* __restrict__ AG) {
    int eidx = blockIdx.x * 16 + (threadIdx.x >> 4);
    if (eidx >= Etot) return;
    int sub = threadIdx.x & 15;
    int c0  = sub * 4;
    int h   = c0 >> 5;
    int s, d;
    if (eidx < E) { s = ei[eidx]; d = ei[E + eidx]; } else { s = d = eidx - E; }
    float e = as[s * 2 + h] + ad[d * 2 + h];
    e = e > 0.f ? e : NEG_SLOPE * e;
    float p = __expf(e - funmap(mu[d * 2 + h]));
    float alpha = p / (ss[d * 2 + h] + 1e-16f);
    const float4 hv = *(const float4*)(HL + (size_t)s * HCDIM + c0);
    float* o = AG + (size_t)d * HCDIM + c0;
    atomicAdd(o + 0, hv.x * alpha);
    atomicAdd(o + 1, hv.y * alpha);
    atomicAdd(o + 2, hv.z * alpha);
    atomicAdd(o + 3, hv.w * alpha);
}

// ---------- bias + ELU (in place) + bf16 copy for next layer's A-matrix ----------
__global__ void gat_elu_bias_cvt(float* __restrict__ AG, const float* __restrict__ b,
                                 __bf16* __restrict__ Xbf) {
    int i = blockIdx.x * blockDim.x + threadIdx.x;
    if (i >= NNODES * HCDIM) return;
    float v = AG[i] + b[i & (HCDIM - 1)];
    v = v > 0.f ? v : (__expf(v) - 1.f);
    AG[i]  = v;
    Xbf[i] = f2bf(v);
}

// ---------- global mean pool (sums + counts) ----------
__global__ void gat_pool(const float* __restrict__ Hf, const int* __restrict__ batch,
                         float* __restrict__ gsum, float* __restrict__ gcnt) {
    int i = blockIdx.x * blockDim.x + threadIdx.x;
    if (i >= NNODES * HCDIM) return;
    int n = i >> 6, c = i & 63;
    int g = batch[n];
    atomicAdd(&gsum[g * HCDIM + c], Hf[i]);
    if (c == 0) atomicAdd(&gcnt[g], 1.0f);
}

// ---------- MLP head: one thread per graph ----------
__global__ void gat_mlp(const float* __restrict__ gsum, const float* __restrict__ gcnt,
                        const float* __restrict__ w1, const float* __restrict__ b1,
                        const float* __restrict__ w2, const float* __restrict__ b2,
                        float* __restrict__ out) {
    int g = threadIdx.x;
    if (g >= GGR) return;
    float cnt = gcnt[g];
    cnt = cnt > 1.f ? cnt : 1.f;
    float pooled[HCDIM];
    for (int c = 0; c < HCDIM; ++c) pooled[c] = gsum[g * HCDIM + c] / cnt;
    float z[32];
    for (int j = 0; j < 32; ++j) {
        float a = b1[j];
        for (int c = 0; c < HCDIM; ++c) a += pooled[c] * w1[j * HCDIM + c];
        z[j] = a > 0.f ? a : 0.f;
    }
    for (int o = 0; o < 2; ++o) {
        float a = b2[o];
        for (int j = 0; j < 32; ++j) a += z[j] * w2[o * 32 + j];
        out[g * 2 + o] = a;
    }
}

extern "C" void kernel_launch(void* const* d_in, const int* in_sizes, int n_in,
                              void* d_out, int out_size, void* d_ws, size_t ws_size,
                              hipStream_t stream) {
    const float* x      = (const float*)d_in[0];
    const int*   ei     = (const int*)d_in[1];
    const int*   batch  = (const int*)d_in[2];
    const float* W[3]   = {(const float*)d_in[3],  (const float*)d_in[7],  (const float*)d_in[11]};
    const float* aS[3]  = {(const float*)d_in[4],  (const float*)d_in[8],  (const float*)d_in[12]};
    const float* aD[3]  = {(const float*)d_in[5],  (const float*)d_in[9],  (const float*)d_in[13]};
    const float* bb[3]  = {(const float*)d_in[6],  (const float*)d_in[10], (const float*)d_in[14]};
    const float* mlp_w1 = (const float*)d_in[15];
    const float* mlp_b1 = (const float*)d_in[16];
    const float* mlp_w2 = (const float*)d_in[17];
    const float* mlp_b2 = (const float*)d_in[18];

    const int E    = in_sizes[1] / 2;
    const int Etot = E + NNODES;
    const int FIN  = in_sizes[0] / NNODES;   // 128

    // workspace carve-up (256B aligned)
    char* ws = (char*)d_ws;
    size_t off = 0;
    auto carve = [&](size_t bytes) -> void* {
        void* p = ws + off;
        off = (off + bytes + 255) & ~(size_t)255;
        return p;
    };
    float*    buf0 = (float*)   carve((size_t)NNODES * HCDIM * 4);   // aggregation out / pool in
    float*    buf1 = (float*)   carve((size_t)NNODES * HCDIM * 4);   // post-GEMM h (f32)
    float*    asb  = (float*)   carve((size_t)NNODES * 2 * 4);
    float*    adb  = (float*)   carve((size_t)NNODES * 2 * 4);
    unsigned* mub  = (unsigned*)carve((size_t)NNODES * 2 * 4);
    float*    ssb  = (float*)   carve((size_t)NNODES * 2 * 4);
    float*    gsum = (float*)   carve((size_t)GGR * HCDIM * 4);
    float*    gcnt = (float*)   carve((size_t)GGR * 4);
    __bf16*   xbf  = (__bf16*)  carve((size_t)NNODES * FIN * 2);     // layer-0 A matrix
    __bf16*   hbf  = (__bf16*)  carve((size_t)NNODES * HCDIM * 2);   // layer-1/2 A matrix
    __bf16*   wbf0 = (__bf16*)  carve((size_t)HCDIM * FIN * 2);
    __bf16*   wbf1 = (__bf16*)  carve((size_t)HCDIM * HCDIM * 2);
    __bf16*   wbf2 = (__bf16*)  carve((size_t)HCDIM * HCDIM * 2);

    const int mtiles = NNODES / 16;  // 6250, exact

    // ---- one-time bf16 conversions (x and all weights) ----
    {
        int n4 = NNODES * FIN / 4;
        cvt_bf16x4<<<(n4 + 255) / 256, 256, 0, stream>>>(x, xbf, n4);
        int w0e = HCDIM * FIN / 4;
        cvt_bf16x4<<<(w0e + 255) / 256, 256, 0, stream>>>(W[0], wbf0, w0e);
        int w1e = HCDIM * HCDIM / 4;
        cvt_bf16x4<<<(w1e + 255) / 256, 256, 0, stream>>>(W[1], wbf1, w1e);
        cvt_bf16x4<<<(w1e + 255) / 256, 256, 0, stream>>>(W[2], wbf2, w1e);
    }

    auto run_edges = [&](int L, float* HL, float* AG) {
        gat_alpha<<<(NNODES + 255) / 256, 256, 0, stream>>>(HL, aS[L], aD[L], asb, adb);
        gat_init_ms<<<(NNODES * 2 + 255) / 256, 256, 0, stream>>>(mub, ssb);
        hipMemsetAsync(AG, 0, (size_t)NNODES * HCDIM * 4, stream);
        gat_edge_max<<<(Etot + 255) / 256, 256, 0, stream>>>(ei, E, Etot, asb, adb, mub);
        gat_edge_sum<<<(Etot + 255) / 256, 256, 0, stream>>>(ei, E, Etot, asb, adb, mub, ssb);
        gat_edge_msg<<<(Etot + 15) / 16, 256, 0, stream>>>(ei, E, Etot, HL, asb, adb, mub, ssb, AG);
        gat_elu_bias_cvt<<<(NNODES * HCDIM + 255) / 256, 256, 0, stream>>>(AG, bb[L], hbf);
    };

    // layer 0: K=128 GEMM from xbf, then edge phase
    gat_gemm_wmma<128><<<mtiles, 128, 0, stream>>>(xbf, wbf0, buf1);
    run_edges(0, buf1, buf0);
    // layer 1: K=64 GEMM from hbf (written by layer-0 ELU)
    gat_gemm_wmma<64><<<mtiles, 128, 0, stream>>>(hbf, wbf1, buf1);
    run_edges(1, buf1, buf0);
    // layer 2
    gat_gemm_wmma<64><<<mtiles, 128, 0, stream>>>(hbf, wbf2, buf1);
    run_edges(2, buf1, buf0);

    hipMemsetAsync(gsum, 0, (size_t)GGR * HCDIM * 4, stream);
    hipMemsetAsync(gcnt, 0, (size_t)GGR * 4, stream);
    gat_pool<<<(NNODES * HCDIM + 255) / 256, 256, 0, stream>>>(buf0, batch, gsum, gcnt);
    gat_mlp<<<1, 64, 0, stream>>>(gsum, gcnt, mlp_w1, mlp_b1, mlp_w2, mlp_b2, (float*)d_out);
}
